// TransformerLayer_43843026158032
// MI455X (gfx1250) — matrix operand order repaired and checked by reading, not compiled, not run on hardware
//
#include <hip/hip_runtime.h>
#include <hip/hip_bf16.h>
#include <math.h>

// ---------------- fixed problem shape ----------------
#define BATCH   2
#define SEQ     2048
#define DM      1024            // d_model
#define NH      16              // heads
#define DK      64              // head dim
#define MROWS   (BATCH * SEQ)   // 4096 GEMM rows

typedef __attribute__((ext_vector_type(16))) __bf16 v16bf;
typedef __attribute__((ext_vector_type(8)))  __bf16 v8bf;
typedef __attribute__((ext_vector_type(4)))  __bf16 v4bf;
typedef __attribute__((ext_vector_type(8)))  float  v8f;

union FragU { v16bf v; v8bf h[2]; };

__device__ __forceinline__ v8f wmma_bf16(v16bf a, v16bf b, v8f c) {
  // D = A(16x32 bf16) x B(32x16 bf16) + C(16x16 f32)
  return __builtin_amdgcn_wmma_f32_16x16x32_bf16(
      /*neg_a=*/false, a, /*neg_b=*/false, b,
      /*c_mod=*/(short)0, c, /*reuse_a=*/false, /*reuse_b=*/false);
}

// A-operand fragment: 16(M) x 32(K), source row-major [m][k], ld elements.
// lane m = l%16 ; elem i -> k = (i&7) + 16*(i>>3) + 8*(l>>4)  (two 16B runs)
__device__ __forceinline__ v16bf load_a(const __bf16* base, int ld, int lane) {
  const int m = lane & 15, hi = lane >> 4;
  const __bf16* p = base + (size_t)m * ld + hi * 8;
  FragU f;
  f.h[0] = *(const v8bf*)(p);
  f.h[1] = *(const v8bf*)(p + 16);
  return f.v;
}

// B-operand fragment: 32(K) x 16(N); source = B^T row-major [n][k], ld elements.
// lane n = l%16 ; elem i -> k = (i&7) + 16*(i>>3) + 8*(l>>4)  (two 16B runs)
__device__ __forceinline__ v16bf load_b(const __bf16* base, int ld, int lane) {
  const int n = lane & 15, hi = lane >> 4;
  const __bf16* p = base + (size_t)n * ld + hi * 8;
  FragU f;
  f.h[0] = *(const v8bf*)(p);
  f.h[1] = *(const v8bf*)(p + 16);
  return f.v;
}

// ---------------- conversion kernels ----------------
__global__ __launch_bounds__(256) void cvt_bf16(const float* __restrict__ in,
                                                __bf16* __restrict__ out, int n) {
  int i = (blockIdx.x * 256 + threadIdx.x) * 4;
  if (i + 3 < n) {
    const float4 v = *(const float4*)(in + i);
    v4bf o;
    o[0] = (__bf16)v.x; o[1] = (__bf16)v.y; o[2] = (__bf16)v.z; o[3] = (__bf16)v.w;
    *(v4bf*)(out + i) = o;
  }
}

// W [DM][DM] f32 row-major -> Wt [DM][DM] bf16 (transposed)
__global__ __launch_bounds__(256) void cvt_tr(const float* __restrict__ W,
                                              __bf16* __restrict__ Wt) {
  int idx = blockIdx.x * 256 + threadIdx.x;   // coalesced read
  int k = idx >> 10, n = idx & (DM - 1);
  Wt[(size_t)n * DM + k] = (__bf16)W[idx];
}

// ---------------- 4096x1024x1024 bf16 GEMM, 128x128 block tile ----------------
// A [M][K] bf16 rm, Bt [N][K] bf16 rm (i.e. B transposed).
// epi 0: obf[m*N+n] = bf16(acc*scale)           (Q / K projection)
// epi 1: Vt store   [b][h][d][s] bf16           (V projection, transposed)
// epi 2: of32[m*N+n] = resid[m*N+n] + acc       (output proj + residual)
__global__ __launch_bounds__(256) void gemm128(const __bf16* __restrict__ A,
                                               const __bf16* __restrict__ Bt,
                                               int epi, float scale,
                                               __bf16* __restrict__ obf,
                                               const float* __restrict__ resid,
                                               float* __restrict__ of32) {
  constexpr int N = DM, K = DM;
  const int lane = threadIdx.x & 31;
  const int w    = threadIdx.x >> 5;         // 8 waves: 4(M) x 2(N)
  const int row0 = blockIdx.y * 128 + (w & 3) * 32;   // wave: 32 rows
  const int col0 = blockIdx.x * 128 + (w >> 2) * 64;  // wave: 64 cols

  v8f acc[2][4] = {};
  for (int kk = 0; kk < K; kk += 32) {
    __builtin_prefetch(A  + (size_t)row0 * K + kk + 128, 0, 1);
    __builtin_prefetch(Bt + (size_t)col0 * K + kk + 128, 0, 1);
    v16bf a0 = load_a(A + (size_t)(row0 +  0) * K + kk, K, lane);
    v16bf a1 = load_a(A + (size_t)(row0 + 16) * K + kk, K, lane);
    v16bf b0 = load_b(Bt + (size_t)(col0 +  0) * K + kk, K, lane);
    v16bf b1 = load_b(Bt + (size_t)(col0 + 16) * K + kk, K, lane);
    v16bf b2 = load_b(Bt + (size_t)(col0 + 32) * K + kk, K, lane);
    v16bf b3 = load_b(Bt + (size_t)(col0 + 48) * K + kk, K, lane);
    acc[0][0] = wmma_bf16(a0, b0, acc[0][0]);
    acc[0][1] = wmma_bf16(a0, b1, acc[0][1]);
    acc[0][2] = wmma_bf16(a0, b2, acc[0][2]);
    acc[0][3] = wmma_bf16(a0, b3, acc[0][3]);
    acc[1][0] = wmma_bf16(a1, b0, acc[1][0]);
    acc[1][1] = wmma_bf16(a1, b1, acc[1][1]);
    acc[1][2] = wmma_bf16(a1, b2, acc[1][2]);
    acc[1][3] = wmma_bf16(a1, b3, acc[1][3]);
  }

  const int hi = lane >> 4, nn = lane & 15;    // C layout: n=l%16, m=r+8*(l>>4)
#pragma unroll
  for (int mi = 0; mi < 2; ++mi)
#pragma unroll
    for (int ni = 0; ni < 4; ++ni)
#pragma unroll
      for (int r = 0; r < 8; ++r) {
        const int m = row0 + mi * 16 + r + 8 * hi;
        const int n = col0 + ni * 16 + nn;
        const float v = acc[mi][ni][r];
        if (epi == 0) {
          obf[(size_t)m * N + n] = (__bf16)(v * scale);
        } else if (epi == 1) {            // Vt: [b][h][d][s]
          const int bb = m >> 11, ss = m & (SEQ - 1);
          const int hh = n >> 6,  dd = n & (DK - 1);
          obf[((((size_t)bb * NH + hh) * DK + dd) << 11) + ss] = (__bf16)v;
        } else {
          const size_t idx = (size_t)m * N + n;
          of32[idx] = resid[idx] + v;
        }
      }
}

// ---------------- causal flash attention: one wave per 16-query tile ----------
// Q [4096][1024] bf16 (pre-scaled by 1/8), Km [4096][1024] bf16,
// Vt [B*H*64][2048] bf16, O [4096][1024] bf16.
__global__ __launch_bounds__(256) void attn(const __bf16* __restrict__ Q,
                                            const __bf16* __restrict__ Km,
                                            const __bf16* __restrict__ Vt,
                                            __bf16* __restrict__ O) {
  __shared__ alignas(128) __bf16 plds[8][16 * 32];   // per-wave P staging (1 KB)
  const int lane = threadIdx.x & 31;
  const int w    = threadIdx.x >> 5;
  const int wid  = blockIdx.x * 8 + w;               // 4096 wave-tiles total
  const int qt = wid & 127;                          // 128 q-tiles of 16
  const int h  = (wid >> 7) & (NH - 1);
  const int b  = wid >> 11;
  const int q0 = qt * 16;

  const __bf16* Qh = Q  + (size_t)b * SEQ * DM + h * DK;
  const __bf16* Kh = Km + (size_t)b * SEQ * DM + h * DK;
  const __bf16* Vh = Vt + ((size_t)(b * NH + h) * DK) * SEQ;

  v16bf qf0 = load_a(Qh + (size_t)q0 * DM +  0, DM, lane);
  v16bf qf1 = load_a(Qh + (size_t)q0 * DM + 32, DM, lane);

  v8f   acc[4] = {};
  float mrow[8], lrow[8];
#pragma unroll
  for (int r = 0; r < 8; ++r) { mrow[r] = -__builtin_inff(); lrow[r] = 0.0f; }

  const int hi = lane >> 4, nn = lane & 15;
  const int jend = q0 + 16;                          // keys 0 .. q0+15
  for (int j = 0; j < jend; j += 32) {               // 32-key blocks
    // ---- scores: two 16x16 tiles (keys j..j+15 and j+16..j+31) ----
    v8f s0 = {}, s1 = {};
    s0 = wmma_bf16(qf0, load_b(Kh + (size_t)(j     ) * DM +  0, DM, lane), s0);
    s0 = wmma_bf16(qf1, load_b(Kh + (size_t)(j     ) * DM + 32, DM, lane), s0);
    s1 = wmma_bf16(qf0, load_b(Kh + (size_t)(j + 16) * DM +  0, DM, lane), s1);
    s1 = wmma_bf16(qf1, load_b(Kh + (size_t)(j + 16) * DM + 32, DM, lane), s1);

    if (j + 31 > q0) {                               // causal mask (edge blocks)
#pragma unroll
      for (int r = 0; r < 8; ++r) {
        const int qrow = q0 + r + 8 * hi;
        if (j + nn      > qrow) s0[r] = -__builtin_inff();
        if (j + 16 + nn > qrow) s1[r] = -__builtin_inff();
      }
    }

    // ---- online softmax (row = 16 lanes of one half-wave) ----
    float pmax[8];
#pragma unroll
    for (int r = 0; r < 8; ++r) pmax[r] = fmaxf(s0[r], s1[r]);
#pragma unroll
    for (int mk = 1; mk < 16; mk <<= 1)
#pragma unroll
      for (int r = 0; r < 8; ++r)
        pmax[r] = fmaxf(pmax[r], __shfl_xor(pmax[r], mk, 32));

    float alpha[8], psum[8];
    v8f p0, p1;
#pragma unroll
    for (int r = 0; r < 8; ++r) {
      const float mn = fmaxf(mrow[r], pmax[r]);
      alpha[r] = __expf(mrow[r] - mn);
      mrow[r]  = mn;
      p0[r] = __expf(s0[r] - mn);
      p1[r] = __expf(s1[r] - mn);
      psum[r] = p0[r] + p1[r];
    }
#pragma unroll
    for (int mk = 1; mk < 16; mk <<= 1)
#pragma unroll
      for (int r = 0; r < 8; ++r)
        psum[r] += __shfl_xor(psum[r], mk, 32);
#pragma unroll
    for (int r = 0; r < 8; ++r) lrow[r] = lrow[r] * alpha[r] + psum[r];
#pragma unroll
    for (int ni = 0; ni < 4; ++ni)
#pragma unroll
      for (int r = 0; r < 8; ++r) acc[ni][r] *= alpha[r];

    // ---- repack P (C layout) -> A layout through per-wave LDS ----
    __bf16* pl = &plds[w][0];
#pragma unroll
    for (int r = 0; r < 8; ++r) {
      const int m = r + 8 * hi;
      pl[m * 32 + nn]      = (__bf16)p0[r];
      pl[m * 32 + 16 + nn] = (__bf16)p1[r];
    }
    const v16bf pf = load_a(pl, 32, lane);           // ds_load_b128 x2, in-order

    // ---- PV: P(16x32) x V(32x64) via Vt (keys contiguous per dim row) ----
#pragma unroll
    for (int ni = 0; ni < 4; ++ni) {
      const v16bf vf = load_b(Vh + (size_t)(ni * 16) * SEQ + j, SEQ, lane);
      acc[ni] = wmma_bf16(pf, vf, acc[ni]);
    }
  }

  // ---- normalize and store merged-head output ----
  float inv[8];
#pragma unroll
  for (int r = 0; r < 8; ++r) inv[r] = 1.0f / lrow[r];
#pragma unroll
  for (int ni = 0; ni < 4; ++ni)
#pragma unroll
    for (int r = 0; r < 8; ++r) {
      const int m = q0 + r + 8 * hi;
      const int n = h * DK + ni * 16 + nn;
      O[((size_t)b * SEQ + m) * DM + n] = (__bf16)(acc[ni][r] * inv[r]);
    }
}

// ---------------- host-side orchestration ----------------
extern "C" void kernel_launch(void* const* d_in, const int* in_sizes, int n_in,
                              void* d_out, int out_size, void* d_ws, size_t ws_size,
                              hipStream_t stream) {
  const float* x  = (const float*)d_in[0];
  const float* Wq = (const float*)d_in[1];
  const float* Wk = (const float*)d_in[2];
  const float* Wv = (const float*)d_in[3];
  const float* Wo = (const float*)d_in[4];
  float* out = (float*)d_out;

  char* ws = (char*)d_ws;
  size_t off = 0;
  auto take = [&](size_t bytes) { char* p = ws + off; off += (bytes + 255) & ~size_t(255); return p; };
  __bf16* xb  = (__bf16*)take((size_t)MROWS * DM * 2);   // x bf16 (reused as AO later)
  __bf16* WqT = (__bf16*)take((size_t)DM * DM * 2);
  __bf16* WkT = (__bf16*)take((size_t)DM * DM * 2);
  __bf16* WvT = (__bf16*)take((size_t)DM * DM * 2);
  __bf16* WoT = (__bf16*)take((size_t)DM * DM * 2);
  __bf16* Qb  = (__bf16*)take((size_t)MROWS * DM * 2);
  __bf16* Kbm = (__bf16*)take((size_t)MROWS * DM * 2);
  __bf16* Vt  = (__bf16*)take((size_t)MROWS * DM * 2);
  __bf16* AO  = xb;   // x bf16 dead after QKV projections -> reuse for attn out

  // 1. precision conversion (+ weight transposes for contiguous B fragments)
  cvt_bf16<<<(MROWS * DM) / (256 * 4), 256, 0, stream>>>(x, xb, MROWS * DM);
  cvt_tr <<<(DM * DM) / 256, 256, 0, stream>>>(Wq, WqT);
  cvt_tr <<<(DM * DM) / 256, 256, 0, stream>>>(Wk, WkT);
  cvt_tr <<<(DM * DM) / 256, 256, 0, stream>>>(Wv, WvT);
  cvt_tr <<<(DM * DM) / 256, 256, 0, stream>>>(Wo, WoT);

  // 2. QKV projections (Q pre-scaled by 1/sqrt(dk)=0.125; V stored transposed)
  dim3 gg(DM / 128, MROWS / 128);
  gemm128<<<gg, 256, 0, stream>>>(xb, WqT, 0, 0.125f, Qb,  nullptr, nullptr);
  gemm128<<<gg, 256, 0, stream>>>(xb, WkT, 0, 1.0f,   Kbm, nullptr, nullptr);
  gemm128<<<gg, 256, 0, stream>>>(xb, WvT, 1, 1.0f,   Vt,  nullptr, nullptr);

  // 3. causal flash attention: 4096 wave-tiles, 8 waves / block
  attn<<<(BATCH * NH * (SEQ / 16)) / 8, 256, 0, stream>>>(Qb, Kbm, Vt, AO);

  // 4. output projection + residual (f32 out)
  gemm128<<<gg, 256, 0, stream>>>(AO, WoT, 2, 1.0f, nullptr, x, out);
}
